// ForwardFromWindow_19138374271107
// MI455X (gfx1250) — matrix operand order repaired and checked by reading, not compiled, not run on hardware
//
#include <hip/hip_runtime.h>
#include <hip/hip_bf16.h>
#include <math.h>

// ---------------------------------------------------------------------------
// CDNA5 (gfx1250) implementation of the BiLSTM seq2seq reference.
// All dense matmuls (gate precompute x@Wx, input_proj, mlp1, output) and the
// recurrent h@Wh use v_wmma_f32_16x16x32_f16 (f16 inputs, f32 accumulate).
// LDS tiles are stored in *fragment-major* layout so every WMMA operand is a
// single 32-byte vector LDS load (2x ds_load_b128), no per-element packing.
//
// Fragment layouts (from CDNA5 ISA 7.12.2, wave32):
//  A (16x32 f16): lane L<16 -> row L,    K={kb..kb+7, 16+kb..16+kb+7}, kb=0
//                 lane L>=16 -> row L-16, same with kb=8
//    element e:  e<8 -> K=kb+e ; e>=8 -> K=16+kb+(e-8)
//  B (32x16 f16): lane L<16 -> col L, K as A with kb=0; lane L>=16 -> col L-16,
//                 K as A with kb=8
//  C/D (16x16 f32): lane col=L&15, VGPR i -> row i + (L<16?0:8)
// ---------------------------------------------------------------------------

typedef __attribute__((ext_vector_type(16))) _Float16 v16h;
typedef __attribute__((ext_vector_type(8)))  float    v8f;

#define LATENT 128
#define HID    64
#define NB     512    // batch
#define WIN    96
#define TOUT   384
#define NGATE  256    // 4*HID

__device__ inline float sigmoidf_(float x) { return 1.0f / (1.0f + expf(-x)); }

// Map (k within 32-chunk) -> (lanegroup, element) for the A fragment.
__device__ inline int a_lgrp(int k) { return (k >> 3) & 1; }
__device__ inline int a_elem(int k) { return ((k & 16) ? 8 : 0) + (k & 7); }

// ======================= generic WMMA GEMM =================================
// C[M,N] = A[M,K] @ W[K,N] + bias[N], fp32 in/out, f16 WMMA inner product.
// Block: 256 threads (8 waves). Tile: 128(M) x 64(N), K-chunks of 32.
// Wave w computes rows [w*16, w*16+16) x all 4 N-subtiles.
#define GTM 128
#define GTN 64
#define GTK 32

__global__ void __launch_bounds__(256) gemm_wmma_f16(
    const float* __restrict__ A, const float* __restrict__ W,
    const float* __restrict__ bias, float* __restrict__ C,
    int M, int N, int K)
{
    // Fragment-major tiles: AF[row][lanegrp] , BF[ntile][lane] ; each a v16h.
    __shared__ v16h AF[GTM * 2];        // 8 KB
    __shared__ v16h BF[4 * 32];         // 4 KB
    _Float16* AF16 = (_Float16*)AF;
    _Float16* BF16 = (_Float16*)BF;

    const int tid  = threadIdx.x;
    const int lane = tid & 31;
    const int wave = tid >> 5;
    const int m0   = blockIdx.y * GTM;
    const int n0   = blockIdx.x * GTN;

    v8f acc[4];
    #pragma unroll
    for (int nt = 0; nt < 4; ++nt)
        #pragma unroll
        for (int i = 0; i < 8; ++i) acc[nt][i] = 0.0f;

    for (int k0 = 0; k0 < K; k0 += GTK) {
        __syncthreads();
        // Stage A tile: 128x32 fp32 -> f16, scattered into fragment layout.
        // float4 loads: 4096 elems / 256 thr = 4 vec4 per thread.
        #pragma unroll
        for (int i = 0; i < 4; ++i) {
            int e4 = (tid + i * 256) * 4;
            int r = e4 >> 5, c = e4 & 31;         // c in {0,4,...,28}
            const float4 v = *(const float4*)&A[(size_t)(m0 + r) * K + (k0 + c)];
            int base = (r * 2 + a_lgrp(c)) * 16 + a_elem(c); // 4 consecutive elems
            AF16[base + 0] = (_Float16)v.x;
            AF16[base + 1] = (_Float16)v.y;
            AF16[base + 2] = (_Float16)v.z;
            AF16[base + 3] = (_Float16)v.w;
        }
        // Stage W tile: 32x64 into B-fragment layout BF[nt][lgrp*16+n15][e=a_elem(k)]
        #pragma unroll
        for (int i = 0; i < 2; ++i) {
            int e4 = (tid + i * 256) * 4;
            int r = e4 >> 6, c = e4 & 63;         // r = k, c = n (4 consecutive)
            const float4 v = *(const float4*)&W[(size_t)(k0 + r) * N + (n0 + c)];
            int lg  = a_lgrp(r);
            int el  = a_elem(r);
            int nt  = c >> 4;
            int n15 = c & 15;                      // n15..n15+3 same tile
            int base = (nt * 32 + lg * 16 + n15) * 16 + el;
            BF16[base + 0 * 16] = (_Float16)v.x;
            BF16[base + 1 * 16] = (_Float16)v.y;
            BF16[base + 2 * 16] = (_Float16)v.z;
            BF16[base + 3 * 16] = (_Float16)v.w;
        }
        __syncthreads();

        const v16h af = AF[(wave * 16 + (lane & 15)) * 2 + ((lane < 16) ? 0 : 1)];
        #pragma unroll
        for (int nt = 0; nt < 4; ++nt) {
            const v16h bf = BF[nt * 32 + lane];
            acc[nt] = __builtin_amdgcn_wmma_f32_16x16x32_f16(
                false, af, false, bf, (short)0, acc[nt], false, false);
        }
    }

    const int col   = lane & 15;
    const int rbase = (lane < 16) ? 0 : 8;
    #pragma unroll
    for (int nt = 0; nt < 4; ++nt) {
        const int n  = n0 + nt * 16 + col;
        const float bv = bias ? bias[n] : 0.0f;
        #pragma unroll
        for (int i = 0; i < 8; ++i) {
            const int m = m0 + wave * 16 + rbase + i;
            C[(size_t)m * N + n] = acc[nt][i] + bv;
        }
    }
}

// ======================= LSTM scan (recurrent part) ========================
// Block: 128 threads (4 waves) handles 16 batch rows for the whole time loop.
// xg: (B,T,256) precomputed x@Wx+b. Wh (64x256) resident in LDS as f16 in
// B-fragment layout. h kept in LDS *in A-fragment f16 layout*, written
// directly by the gate-update phase, so each step's operands are single
// vector LDS loads.
#define GSPAD 260

__global__ void __launch_bounds__(128) lstm_scan(
    const float* __restrict__ xg, const float* __restrict__ Wh,
    float* __restrict__ ys, int ys_stride, int ys_off,
    float* __restrict__ hfin, int hf_stride, int hf_off,
    int T, int reverse)
{
    __shared__ v16h WhF[2 * 16 * 32];   // [kchunk][ntile][lane] : 32 KB
    __shared__ v16h hF[2 * 16 * 2];     // [kchunk][row][lanegrp] : 2 KB
    __shared__ float gs[16 * GSPAD];    // gate pre-activations (~16.6 KB)
    __shared__ float cs[16 * HID];      // cell state (4 KB)
    _Float16* WhF16 = (_Float16*)WhF;
    _Float16* hF16  = (_Float16*)hF;

    const int tid  = threadIdx.x;
    const int lane = tid & 31;
    const int wave = tid >> 5;
    const int b0   = blockIdx.x * 16;

    // Stage Wh (64x256) into B-fragment layout: WhF[kc][nt][lgrp*16+n15][a_elem(k)]
    for (int e = tid; e < 64 * 256; e += 128) {
        int r = e >> 8, c = e & 255;               // r = k (0..63), c = n
        int kc  = r >> 5;
        int k5  = r & 31;
        int lg  = a_lgrp(k5);
        int el  = a_elem(k5);
        int nt  = c >> 4;
        int n15 = c & 15;
        WhF16[(((kc * 16 + nt) * 32) + lg * 16 + n15) * 16 + el] = (_Float16)Wh[e];
    }
    for (int e = tid; e < 2 * 16 * 2 * 16; e += 128) hF16[e] = (_Float16)0.0f;
    for (int e = tid; e < 16 * HID; e += 128) cs[e] = 0.0f;
    __syncthreads();

    const int urow = tid >> 3;        // state-update row (0..15)
    const int un0  = (tid & 7) * 8;   // state-update col base (8 cols/thread)
    const int n15  = lane & 15;
    const int lg   = (lane < 16) ? 0 : 1;
    const int rb   = (lane < 16) ? 0 : 8;

    for (int s = 0; s < T; ++s) {
        const int t = reverse ? (T - 1 - s) : s;

        // A fragments: single vector LDS loads (h already in frag layout)
        const v16h a0 = hF[(0 * 16 + n15) * 2 + lg];
        const v16h a1 = hF[(1 * 16 + n15) * 2 + lg];
        #pragma unroll
        for (int nt4 = 0; nt4 < 4; ++nt4) {
            const int nt = wave * 4 + nt4;
            const v16h bf0 = WhF[(0 * 16 + nt) * 32 + lane];
            const v16h bf1 = WhF[(1 * 16 + nt) * 32 + lane];
            v8f acc;
            #pragma unroll
            for (int i = 0; i < 8; ++i) acc[i] = 0.0f;
            acc = __builtin_amdgcn_wmma_f32_16x16x32_f16(
                false, a0, false, bf0, (short)0, acc, false, false);
            acc = __builtin_amdgcn_wmma_f32_16x16x32_f16(
                false, a1, false, bf1, (short)0, acc, false, false);
            const int n = nt * 16 + n15;
            #pragma unroll
            for (int i = 0; i < 8; ++i) {
                const int r = rb + i;
                gs[r * GSPAD + n] =
                    acc[i] + xg[((size_t)(b0 + r) * T + t) * NGATE + n];
            }
        }
        __syncthreads();

        // Gate nonlinearity + state update (i,f,g,o at n, 64+n, 128+n, 192+n).
        // Writes h as f16 directly into A-fragment layout (8 contiguous halves).
        {
            const int kc  = un0 >> 5;
            const int k5  = un0 & 31;
            const int hbase = ((kc * 16 + urow) * 2 + a_lgrp(k5)) * 16 + a_elem(k5);
            #pragma unroll
            for (int j = 0; j < 8; ++j) {
                const int n  = un0 + j;
                const float ig = sigmoidf_(gs[urow * GSPAD + n]);
                const float fg = sigmoidf_(gs[urow * GSPAD + 64 + n]);
                const float gg = tanhf(gs[urow * GSPAD + 128 + n]);
                const float og = sigmoidf_(gs[urow * GSPAD + 192 + n]);
                const float c  = fg * cs[urow * HID + n] + ig * gg;
                cs[urow * HID + n] = c;
                const float h  = og * tanhf(c);
                hF16[hbase + j] = (_Float16)h;
                if (ys)
                    ys[((size_t)(b0 + urow) * T + t) * ys_stride + ys_off + n] = h;
                if (hfin && s == T - 1)
                    hfin[(size_t)(b0 + urow) * hf_stride + hf_off + n] = h;
            }
        }
        __syncthreads();
    }
}

// ================= input_proj: x = [h_rep, pos] @ W + b ====================
// h_rep is time-constant, so base = h_init@W[0:128]+b is one small GEMM;
// this kernel adds the rank-1 pos term: x[b,t,n] = base[b,n] + (t/(T-1))*W[128,n]
__global__ void posproj_kernel(const float* __restrict__ base,
                               const float* __restrict__ Wp,   // (129,128)
                               float* __restrict__ x2, int T)
{
    const int idx = blockIdx.x * blockDim.x + threadIdx.x;
    const int total = NB * TOUT * LATENT;
    if (idx >= total) return;
    const int n = idx & 127;
    const int t = (idx >> 7) % T;
    const int b = idx / (T * LATENT);
    const float pos = (float)t / (float)(T - 1);
    x2[idx] = base[b * LATENT + n] + pos * Wp[128 * LATENT + n];
}

// ==================== fused LayerNorm + gelu (tanh) ========================
__global__ void __launch_bounds__(128) ln_gelu_kernel(
    float* __restrict__ H, const float* __restrict__ g,
    const float* __restrict__ b)
{
    __shared__ float red[128];
    const size_t row = blockIdx.x;
    const int n = threadIdx.x;
    const float x = H[row * 128 + n];

    red[n] = x;
    __syncthreads();
    for (int s = 64; s > 0; s >>= 1) {
        if (n < s) red[n] += red[n + s];
        __syncthreads();
    }
    const float mu = red[0] * (1.0f / 128.0f);
    __syncthreads();
    const float d = x - mu;
    red[n] = d * d;
    __syncthreads();
    for (int s = 64; s > 0; s >>= 1) {
        if (n < s) red[n] += red[n + s];
        __syncthreads();
    }
    const float var = red[0] * (1.0f / 128.0f);
    float y = d * rsqrtf(var + 1e-6f) * g[n] + b[n];
    // jax.nn.gelu (approximate=True)
    const float k0 = 0.7978845608028654f;
    const float y3 = y * y * y;
    H[row * 128 + n] = 0.5f * y * (1.0f + tanhf(k0 * (y + 0.044715f * y3)));
}

// ============================ host launcher ================================
static inline void launch_gemm(const float* A, const float* W, const float* bias,
                               float* C, int M, int N, int K, hipStream_t s)
{
    dim3 grid(N / GTN, M / GTM);
    gemm_wmma_f16<<<grid, 256, 0, s>>>(A, W, bias, C, M, N, K);
}

extern "C" void kernel_launch(void* const* d_in, const int* in_sizes, int n_in,
                              void* d_out, int out_size, void* d_ws, size_t ws_size,
                              hipStream_t stream) {
    // Inputs in setup_inputs() order (params depth-first in dict insertion order)
    const float* z        = (const float*)d_in[0];   // (512,96,128)
    // d_in[1] = T_out scalar (==384, baked in)
    const float* e1f0_Wx  = (const float*)d_in[2];
    const float* e1f0_Wh  = (const float*)d_in[3];
    const float* e1f0_b   = (const float*)d_in[4];
    const float* e1b0_Wx  = (const float*)d_in[5];
    const float* e1b0_Wh  = (const float*)d_in[6];
    const float* e1b0_b   = (const float*)d_in[7];
    const float* e1f1_Wx  = (const float*)d_in[8];
    const float* e1f1_Wh  = (const float*)d_in[9];
    const float* e1f1_b   = (const float*)d_in[10];
    const float* e1b1_Wx  = (const float*)d_in[11];
    const float* e1b1_Wh  = (const float*)d_in[12];
    const float* e1b1_b   = (const float*)d_in[13];
    const float* ip_W     = (const float*)d_in[14];  // (129,128)
    const float* ip_b     = (const float*)d_in[15];
    const float* e2f_Wx   = (const float*)d_in[16];
    const float* e2f_Wh   = (const float*)d_in[17];
    const float* e2f_b    = (const float*)d_in[18];
    const float* e2b_Wx   = (const float*)d_in[19];
    const float* e2b_Wh   = (const float*)d_in[20];
    const float* e2b_b    = (const float*)d_in[21];
    const float* dec_Wx   = (const float*)d_in[22];
    const float* dec_Wh   = (const float*)d_in[23];
    const float* dec_b    = (const float*)d_in[24];
    const float* m1_W     = (const float*)d_in[25];  // (64,128)
    const float* m1_b     = (const float*)d_in[26];
    const float* ln_g     = (const float*)d_in[27];
    const float* ln_b     = (const float*)d_in[28];
    const float* out_W    = (const float*)d_in[29];  // (128,128)
    const float* out_b    = (const float*)d_in[30];

    float* out = (float*)d_out;

    // Workspace layout (fp32 elements)
    float* ws = (float*)d_ws;
    const size_t SZ_XG = (size_t)NB * TOUT * NGATE;   // 50,331,648
    const size_t SZ_X  = (size_t)NB * TOUT * LATENT;  // 25,165,824
    float* bufA  = ws;                       // xg fwd
    float* bufB  = bufA + SZ_XG;             // xg bwd
    float* bufX  = bufB + SZ_XG;             // (B,T,128) activations
    float* bufY  = bufX + SZ_X;              // (B,T,128) activations
    float* base  = bufY + SZ_X;              // (B,128)
    float* hinit = base + (size_t)NB * LATENT;

    const int MB_WIN = NB * WIN;   // 49152
    const int MB_T   = NB * TOUT;  // 196608
    const int scan_blocks = NB / 16;

    // ---- enc1 layer 0 (bidirectional) ----
    launch_gemm(z, e1f0_Wx, e1f0_b, bufA, MB_WIN, NGATE, LATENT, stream);
    launch_gemm(z, e1b0_Wx, e1b0_b, bufB, MB_WIN, NGATE, LATENT, stream);
    lstm_scan<<<scan_blocks, 128, 0, stream>>>(bufA, e1f0_Wh, bufY, 128, 0,
                                               nullptr, 0, 0, WIN, 0);
    lstm_scan<<<scan_blocks, 128, 0, stream>>>(bufB, e1b0_Wh, bufY, 128, 64,
                                               nullptr, 0, 0, WIN, 1);
    // ---- enc1 layer 1: only final hidden states needed ----
    launch_gemm(bufY, e1f1_Wx, e1f1_b, bufA, MB_WIN, NGATE, LATENT, stream);
    launch_gemm(bufY, e1b1_Wx, e1b1_b, bufB, MB_WIN, NGATE, LATENT, stream);
    lstm_scan<<<scan_blocks, 128, 0, stream>>>(bufA, e1f1_Wh, nullptr, 0, 0,
                                               hinit, 128, 0, WIN, 0);
    lstm_scan<<<scan_blocks, 128, 0, stream>>>(bufB, e1b1_Wh, nullptr, 0, 0,
                                               hinit, 128, 64, WIN, 1);

    // ---- input_proj: base = h_init @ W[0:128] + b, then + pos * W[128] ----
    launch_gemm(hinit, ip_W, ip_b, base, NB, LATENT, LATENT, stream);
    {
        const int total = NB * TOUT * LATENT;
        posproj_kernel<<<(total + 255) / 256, 256, 0, stream>>>(base, ip_W,
                                                                bufX, TOUT);
    }

    // ---- enc2 BiLSTM over (B,384,128) ----
    launch_gemm(bufX, e2f_Wx, e2f_b, bufA, MB_T, NGATE, LATENT, stream);
    launch_gemm(bufX, e2b_Wx, e2b_b, bufB, MB_T, NGATE, LATENT, stream);
    lstm_scan<<<scan_blocks, 128, 0, stream>>>(bufA, e2f_Wh, bufY, 128, 0,
                                               nullptr, 0, 0, TOUT, 0);
    lstm_scan<<<scan_blocks, 128, 0, stream>>>(bufB, e2b_Wh, bufY, 128, 64,
                                               nullptr, 0, 0, TOUT, 1);

    // ---- decoder LSTM ----
    launch_gemm(bufY, dec_Wx, dec_b, bufA, MB_T, NGATE, LATENT, stream);
    lstm_scan<<<scan_blocks, 128, 0, stream>>>(bufA, dec_Wh, bufX, 64, 0,
                                               nullptr, 0, 0, TOUT, 0);

    // ---- head: mlp1 -> LN+gelu -> output ----
    launch_gemm(bufX, m1_W, m1_b, bufB, MB_T, LATENT, HID, stream);
    ln_gelu_kernel<<<MB_T, 128, 0, stream>>>(bufB, ln_g, ln_b);
    launch_gemm(bufB, out_W, out_b, out, MB_T, LATENT, LATENT, stream);
}
